// LeafBlockAttention_35467839930525
// MI455X (gfx1250) — matrix operand order repaired and checked by reading, not compiled, not run on hardware
//
#include <hip/hip_runtime.h>
#include <math.h>

// ---- problem constants (from reference) ----
#define B_   2
#define N_   32768
#define C_   512
#define L_   64
#define H_   8
#define DH_  64
#define NB_  512
#define EG_  16
#define C3_  1536   // 3*C

typedef __attribute__((ext_vector_type(16))) __bf16 v16bf;
typedef __attribute__((ext_vector_type(8)))  float  v8f;

// LDS leading dims (padded to dodge 64-bank conflicts)
#define LDX_ (C_ + 4)    // 516
#define LDQ_ (DH_ + 4)   // 68

// LDS layout in floats
#define OFF_XS    0
#define OFF_XMID  (OFF_XS   + L_*LDX_)
#define OFF_QS    (OFF_XMID + L_*LDX_)
#define OFF_KS    (OFF_QS   + L_*LDQ_)
#define OFF_VS    (OFF_KS   + L_*LDQ_)
#define OFF_MEAN  (OFF_VS   + L_*LDQ_)
#define OFF_VBLK  (OFF_MEAN + C_)
#define OFF_GBL   (OFF_VBLK + C_)
#define OFF_WEG1  (OFF_GBL  + L_*H_)
#define OFF_BEG1  (OFF_WEG1 + 4*EG_)
#define OFF_WEG2  (OFF_BEG1 + EG_)
#define OFF_BEG2  (OFF_WEG2 + EG_*H_)
#define SMEM_F    (OFF_BEG2 + H_)
#define SMEM_BYTES (SMEM_F * 4)   // 323,424 B < 320 KB WGP LDS

#if defined(__gfx1250__)
#define ASYNC_X_LOAD 1
#else
#define ASYNC_X_LOAD 0
#endif

// ---------------- WMMA fragment helpers (bf16 in, f32 acc) ----------------
// A: 16(M) x 32(K) tile at p0[m*ld + k].
// Layout: lane<16 -> M=lane, K=e; lane>=16 -> M=lane-16, K=16+e.
__device__ __forceinline__ v16bf load_a_f32(const float* p0, int ld, int lane) {
  const float* p = p0 + (lane & 15) * ld + ((lane >> 4) << 4);
  v16bf a;
#pragma unroll
  for (int e = 0; e < 16; ++e) a[e] = (__bf16)p[e];
  return a;
}
// B: 32(K) x 16(N) tile at p0[k*ld + n].
// Layout (contiguous k, same as load_a_f32): lane<16 -> N=lane, K=e; lane>=16 -> N=lane-16, K=16+e.
__device__ __forceinline__ v16bf load_b_f32(const float* p0, int ld, int lane) {
  const float* p = p0 + (((lane >> 4) << 4)) * ld + (lane & 15);
  v16bf b;
#pragma unroll
  for (int e = 0; e < 16; ++e) b[e] = (__bf16)p[e * ld];
  return b;
}
// B from transposed source: logical B(k,n) = src[n*ld + k]  (for Q*K^T)
__device__ __forceinline__ v16bf load_bt_f32(const float* p0, int ld, int lane) {
  const float* p = p0 + (lane & 15) * ld + ((lane >> 4) << 4);
  v16bf b;
#pragma unroll
  for (int e = 0; e < 16; ++e) b[e] = (__bf16)p[e];
  return b;
}
// D/C 16x16 f32: VGPR r, lane l -> M = r + 8*(l>=16), N = l&15
__device__ __forceinline__ void store_d(float* p0, int ld, int lane, const v8f& d) {
  float* p = p0 + (((lane >> 4) << 3)) * ld + (lane & 15);
#pragma unroll
  for (int r = 0; r < 8; ++r) p[r * ld] = d[r];
}
__device__ __forceinline__ v8f wmma_bf16(v16bf a, v16bf b, v8f c) {
  return __builtin_amdgcn_wmma_f32_16x16x32_bf16(false, a, false, b, (short)0, c,
                                                 false, false);
}

// ---------------- fused per-block kernel ----------------
__global__ void __launch_bounds__(256)
leaf_block_attention_fused(const float* __restrict__ x,
                           const int*   __restrict__ attn_mask,
                           const float* __restrict__ edge,
                           const float* __restrict__ w_qkv,
                           const float* __restrict__ b_qkv,
                           const float* __restrict__ w_proj,
                           const float* __restrict__ b_proj,
                           const float* __restrict__ w_eg1,
                           const float* __restrict__ b_eg1,
                           const float* __restrict__ w_eg2,
                           const float* __restrict__ b_eg2,
                           const float* __restrict__ w_blk,
                           const float* __restrict__ b_blk,
                           float* __restrict__ out) {
  const int blk  = blockIdx.x;      // 0 .. B*NB-1
  const int bb   = blk / NB_;
  const int nb   = blk % NB_;
  const int t    = threadIdx.x;
  const int lane = t & 31;
  const int wave = t >> 5;

  extern __shared__ float sm[];
  float* xs   = sm + OFF_XS;     // 64 x 516 f32 x-tile
  float* xmid = sm + OFF_XMID;   // 64 x 516 f32 attention output (pre-proj)
  float* qs   = sm + OFF_QS;     // 64 x 68  (Q, later re-used for S/probs)
  float* ks   = sm + OFF_KS;     // 64 x 68
  float* vs   = sm + OFF_VS;     // 64 x 68
  float* meanv= sm + OFF_MEAN;   // 512
  float* vblk = sm + OFF_VBLK;   // 512 (block-node V)
  float* gbl  = sm + OFF_GBL;    // 64 x 8 sigmoid gates
  float* weg1 = sm + OFF_WEG1;   // 4 x 16
  float* beg1 = sm + OFF_BEG1;   // 16
  float* weg2 = sm + OFF_WEG2;   // 16 x 8
  float* beg2 = sm + OFF_BEG2;   // 8

  const float* xg = x + ((size_t)bb * N_ + (size_t)nb * L_) * C_;

  // -------- Phase A: stage x tile (64x512 f32) into LDS --------
#if ASYNC_X_LOAD
  // CDNA5 direct global->LDS async copy (GLOBAL_LOAD_ASYNC_TO_LDS_B128,
  // ASYNCcnt-tracked, bypasses VGPRs).  Per-lane LDS destination lets us
  // keep the bank-conflict padding (LDX_=516) that a single TDM descriptor
  // cannot express (pad_interval tops out at 256 DWORDs < 512-DWORD rows).
  // No static __shared__ in this TU, so dynamic-LDS base is 0 and LDS byte
  // offsets equal offsets into sm[].  16B alignment holds on both sides.
  for (int i = 0; i < 32; ++i) {
    int v  = i * 256 + t;        // float4 index, 8192 total
    int r  = v >> 7;             // 128 float4 per row
    int c4 = v & 127;
    unsigned long long ga =
        (unsigned long long)(size_t)(const void*)(xg + r * C_ + c4 * 4);
    unsigned lofs = (unsigned)((OFF_XS + r * LDX_ + c4 * 4) * sizeof(float));
    asm volatile("global_load_async_to_lds_b128 %0, %1, off"
                 :: "v"(lofs), "v"(ga)
                 : "memory");
  }
#else
#pragma unroll 4
  for (int i = 0; i < 32; ++i) {
    int v  = i * 256 + t;        // float4 index, 8192 total
    int r  = v >> 7;             // 128 float4 per row
    int c4 = v & 127;
    const float4 f = *(const float4*)(xg + r * C_ + c4 * 4);
    float* d = xs + r * LDX_ + c4 * 4;
    d[0] = f.x; d[1] = f.y; d[2] = f.z; d[3] = f.w;
  }
#endif
  if (t < 4 * EG_)  weg1[t] = w_eg1[t];
  if (t < EG_)      beg1[t] = b_eg1[t];
  if (t < EG_ * H_) weg2[t] = w_eg2[t];
  if (t < H_)       beg2[t] = b_eg2[t];
#if ASYNC_X_LOAD
  asm volatile("s_wait_asynccnt 0x0" ::: "memory");
#endif
  __syncthreads();

  // -------- Phase B: block mean, g_block gates --------
  for (int c = t; c < C_; c += 256) {
    float s = 0.f;
    for (int r = 0; r < L_; ++r) s += xs[r * LDX_ + c];
    meanv[c] = s * (1.0f / (float)L_);
  }
  for (int i = t; i < L_ * H_; i += 256) {
    int r = i >> 3, hh = i & 7;
    float s = b_blk[hh];
    for (int c = 0; c < C_; ++c) s += xs[r * LDX_ + c] * w_blk[c * H_ + hh];
    gbl[i] = 1.0f / (1.0f + expf(-s));
  }
  __syncthreads();
  // v_block = (mean @ w_qkv + b_qkv)[2C:]
  for (int c = t; c < C_; c += 256) {
    float s = b_qkv[2 * C_ + c];
    for (int k = 0; k < C_; ++k) s += meanv[k] * w_qkv[k * C3_ + 2 * C_ + c];
    vblk[c] = s;
  }

  const float scale = 0.125f;  // DH^-0.5
  const int*   mbase = attn_mask + (nb * L_) * L_;
  const float* ebase = edge + ((size_t)(nb * L_) * L_) * 4;

  // -------- Phase C: per-head attention --------
  for (int h = 0; h < H_; ++h) {
    // C1: Q,K,V (each 64x64) = xs @ w_qkv columns.  12 column-strips of 16.
    for (int cs = wave; cs < 12; cs += 8) {
      int mat = cs >> 2;                       // 0=q,1=k,2=v
      int n0  = (cs & 3) << 4;
      int j0  = mat * C_ + h * DH_ + n0;       // absolute column in w_qkv
      v8f a0 = {}, a1 = {}, a2 = {}, a3 = {};
      for (int kk = 0; kk < C_; kk += 32) {
        v16bf bf = load_b_f32(w_qkv + (size_t)kk * C3_ + j0, C3_, lane);
        v16bf f0 = load_a_f32(xs + (0 * 16) * LDX_ + kk, LDX_, lane);
        v16bf f1 = load_a_f32(xs + (1 * 16) * LDX_ + kk, LDX_, lane);
        v16bf f2 = load_a_f32(xs + (2 * 16) * LDX_ + kk, LDX_, lane);
        v16bf f3 = load_a_f32(xs + (3 * 16) * LDX_ + kk, LDX_, lane);
        a0 = wmma_bf16(f0, bf, a0);
        a1 = wmma_bf16(f1, bf, a1);
        a2 = wmma_bf16(f2, bf, a2);
        a3 = wmma_bf16(f3, bf, a3);
      }
      float bias = b_qkv[j0 + (lane & 15)];
#pragma unroll
      for (int r = 0; r < 8; ++r) { a0[r] += bias; a1[r] += bias; a2[r] += bias; a3[r] += bias; }
      float* dst = (mat == 0) ? qs : (mat == 1) ? ks : vs;
      store_d(dst + (0 * 16) * LDQ_ + n0, LDQ_, lane, a0);
      store_d(dst + (1 * 16) * LDQ_ + n0, LDQ_, lane, a1);
      store_d(dst + (2 * 16) * LDQ_ + n0, LDQ_, lane, a2);
      store_d(dst + (3 * 16) * LDQ_ + n0, LDQ_, lane, a3);
    }
    __syncthreads();

    // C2: S = Q K^T (64x64, 16 tiles; 2 per wave) kept in registers
    const int t0 = wave * 2, t1 = t0 + 1;
    v8f s0 = {}, s1 = {};
    for (int kk = 0; kk < DH_; kk += 32) {
      v16bf qa0 = load_a_f32(qs + ((t0 >> 2) * 16) * LDQ_ + kk, LDQ_, lane);
      v16bf kb0 = load_bt_f32(ks + ((t0 & 3) * 16) * LDQ_ + kk, LDQ_, lane);
      s0 = wmma_bf16(qa0, kb0, s0);
      v16bf qa1 = load_a_f32(qs + ((t1 >> 2) * 16) * LDQ_ + kk, LDQ_, lane);
      v16bf kb1 = load_bt_f32(ks + ((t1 & 3) * 16) * LDQ_ + kk, LDQ_, lane);
      s1 = wmma_bf16(qa1, kb1, s1);
    }
    __syncthreads();  // all waves done reading Q -> reuse qs as score buffer
    store_d(qs + ((t0 >> 2) * 16) * LDQ_ + ((t0 & 3) * 16), LDQ_, lane, s0);
    store_d(qs + ((t1 >> 2) * 16) * LDQ_ + ((t1 & 3) * 16), LDQ_, lane, s1);
    __syncthreads();

    // C3: mask / diag-fix / bias / softmax / edge-gate (4 lanes per row)
    {
      int r  = t >> 2;
      int c0 = (t & 3) << 4;
      const int*   mrow = mbase + r * L_;
      const float* erow = ebase + (size_t)r * L_ * 4;
      int mloc[16];
      int msum = 0;
#pragma unroll
      for (int j = 0; j < 16; ++j) { mloc[j] = mrow[c0 + j]; msum += mloc[j]; }
      msum += __shfl_xor(msum, 1);
      msum += __shfl_xor(msum, 2);
      const int need = (msum < 1) ? 1 : 0;
      float* srow = qs + r * LDQ_;
      float sv[16];
      float mx = -__builtin_inff();
#pragma unroll
      for (int j = 0; j < 16; ++j) {
        int c = c0 + j;
        int m = mloc[j];
        if (c == r && need) m = 1;           // diag = max(diag, need_diag)
        mloc[j] = m;
        float bias3 = (c == r) ? 1.0f : erow[(size_t)c * 4 + 3];
        float s = srow[c] * scale + bias3;
        if (m == 0) s = -__builtin_inff();
        sv[j] = s;
        mx = fmaxf(mx, s);
      }
      mx = fmaxf(mx, __shfl_xor(mx, 1));
      mx = fmaxf(mx, __shfl_xor(mx, 2));
      float sum = 0.f;
#pragma unroll
      for (int j = 0; j < 16; ++j) { sv[j] = expf(sv[j] - mx); sum += sv[j]; }
      sum += __shfl_xor(sum, 1);
      sum += __shfl_xor(sum, 2);
      const float inv = 1.0f / sum;
#pragma unroll
      for (int j = 0; j < 16; ++j) {
        int c = c0 + j;
        float e0, e1, e2, e3;
        if (c == r) { e0 = e1 = e2 = 0.f; e3 = 1.f; }
        else { const float* ep = erow + (size_t)c * 4; e0 = ep[0]; e1 = ep[1]; e2 = ep[2]; e3 = ep[3]; }
        float g = beg2[h];
#pragma unroll
        for (int jj = 0; jj < EG_; ++jj) {
          float hd = beg1[jj] + e0 * weg1[jj] + e1 * weg1[EG_ + jj]
                   + e2 * weg1[2 * EG_ + jj] + e3 * weg1[3 * EG_ + jj];
          hd = 0.5f * hd * (1.0f + erff(hd * 0.70710678118654752f)); // exact gelu
          g += hd * weg2[jj * H_ + h];
        }
        float p = (mloc[j] == 0) ? 0.f : sv[j] * inv;
        if (mloc[j] == 0) g = 0.f;
        srow[c] = p + g;                      // combined = probs + gate
      }
    }
    __syncthreads();

    // C4: O = combined @ V  -> xmid[:, h*64 + ...]
    {
      v8f o0 = {}, o1 = {};
      for (int kk = 0; kk < L_; kk += 32) {
        v16bf pa0 = load_a_f32(qs + ((t0 >> 2) * 16) * LDQ_ + kk, LDQ_, lane);
        v16bf vb0 = load_b_f32(vs + kk * LDQ_ + ((t0 & 3) * 16), LDQ_, lane);
        o0 = wmma_bf16(pa0, vb0, o0);
        v16bf pa1 = load_a_f32(qs + ((t1 >> 2) * 16) * LDQ_ + kk, LDQ_, lane);
        v16bf vb1 = load_b_f32(vs + kk * LDQ_ + ((t1 & 3) * 16), LDQ_, lane);
        o1 = wmma_bf16(pa1, vb1, o1);
      }
      store_d(xmid + ((t0 >> 2) * 16) * LDX_ + h * DH_ + ((t0 & 3) * 16), LDX_, lane, o0);
      store_d(xmid + ((t1 >> 2) * 16) * LDX_ + h * DH_ + ((t1 & 3) * 16), LDX_, lane, o1);
    }
    __syncthreads();
  }  // head loop

  // -------- Phase D: x_mid += g_block * v_block --------
#pragma unroll 4
  for (int i = 0; i < 128; ++i) {
    int v = i * 256 + t;
    int r = v >> 9;
    int c = v & 511;
    xmid[r * LDX_ + c] += gbl[r * H_ + (c >> 6)] * vblk[c];
  }
  __syncthreads();

  // -------- Phase E: out = x_mid @ w_proj + b_proj --------
  float* og = out + ((size_t)bb * N_ + (size_t)nb * L_) * C_;
  for (int cs = wave; cs < 32; cs += 8) {
    int n0 = cs << 4;
    v8f a0 = {}, a1 = {}, a2 = {}, a3 = {};
    for (int kk = 0; kk < C_; kk += 32) {
      v16bf bf = load_b_f32(w_proj + (size_t)kk * C_ + n0, C_, lane);
      v16bf f0 = load_a_f32(xmid + (0 * 16) * LDX_ + kk, LDX_, lane);
      v16bf f1 = load_a_f32(xmid + (1 * 16) * LDX_ + kk, LDX_, lane);
      v16bf f2 = load_a_f32(xmid + (2 * 16) * LDX_ + kk, LDX_, lane);
      v16bf f3 = load_a_f32(xmid + (3 * 16) * LDX_ + kk, LDX_, lane);
      a0 = wmma_bf16(f0, bf, a0);
      a1 = wmma_bf16(f1, bf, a1);
      a2 = wmma_bf16(f2, bf, a2);
      a3 = wmma_bf16(f3, bf, a3);
    }
    float bias = b_proj[n0 + (lane & 15)];
#pragma unroll
    for (int r = 0; r < 8; ++r) { a0[r] += bias; a1[r] += bias; a2[r] += bias; a3[r] += bias; }
    store_d(og + (0 * 16) * C_ + n0, C_, lane, a0);
    store_d(og + (1 * 16) * C_ + n0, C_, lane, a1);
    store_d(og + (2 * 16) * C_ + n0, C_, lane, a2);
    store_d(og + (3 * 16) * C_ + n0, C_, lane, a3);
  }
}

extern "C" void kernel_launch(void* const* d_in, const int* in_sizes, int n_in,
                              void* d_out, int out_size, void* d_ws, size_t ws_size,
                              hipStream_t stream) {
  (void)in_sizes; (void)n_in; (void)out_size; (void)d_ws; (void)ws_size;
  const float* x         = (const float*)d_in[0];
  const int*   attn_mask = (const int*)d_in[1];
  const float* edge      = (const float*)d_in[2];
  const float* w_qkv     = (const float*)d_in[3];
  const float* b_qkv     = (const float*)d_in[4];
  const float* w_proj    = (const float*)d_in[5];
  const float* b_proj    = (const float*)d_in[6];
  const float* w_eg1     = (const float*)d_in[7];
  const float* b_eg1     = (const float*)d_in[8];
  const float* w_eg2     = (const float*)d_in[9];
  const float* b_eg2     = (const float*)d_in[10];
  const float* w_blk     = (const float*)d_in[11];
  const float* b_blk     = (const float*)d_in[12];
  float* out = (float*)d_out;

  hipFuncSetAttribute(reinterpret_cast<const void*>(leaf_block_attention_fused),
                      hipFuncAttributeMaxDynamicSharedMemorySize, SMEM_BYTES);
  leaf_block_attention_fused<<<B_ * NB_, 256, SMEM_BYTES, stream>>>(
      x, attn_mask, edge, w_qkv, b_qkv, w_proj, b_proj,
      w_eg1, b_eg1, w_eg2, b_eg2, w_blk, b_blk, out);
}
